// SoilMoisturePredictorSNN_86655260164622
// MI455X (gfx1250) — matrix operand order repaired and checked by reading, not compiled, not run on hardware
//
#include <hip/hip_runtime.h>

typedef __attribute__((ext_vector_type(16))) _Float16 v16h;
typedef __attribute__((ext_vector_type(8)))  _Float16 v8h;
typedef __attribute__((ext_vector_type(8)))  float    v8f;
typedef __attribute__((ext_vector_type(4)))  float    fvec4;
typedef unsigned int v4u  __attribute__((ext_vector_type(4)));
typedef int          v4i  __attribute__((ext_vector_type(4)));
typedef int          v8i_ __attribute__((ext_vector_type(8)));

#define TILE_B     16    // batch rows per block (WMMA M)
#define WAVES      8     // waves per block; each owns 32 hidden cols
#define NT         2     // 16-wide N tiles per wave (8*32 = 256 cols)
#define S_DIM      64    // sensors (WMMA K, split 2 x 32)
#define H_DIM      256
#define T_DIM      1024
#define B_DIM      2048

// exp(-1/50), exp(-1/30)
#define BETA_H 0.98019867f
#define BETA_O 0.96721610f

// ---- Tensor Data Mover: 2-D tile load (16 rows x 64 f32, row stride T*S) ----
// D# per CDNA5 ISA 8.3/8.4: group0 = {count|flags, lds_addr, gaddr_lo,
// gaddr_hi|type=2}; group1 = {data_size=4B, tensor_dim0=64, tensor_dim1=16,
// tile_dim0=64, tile_dim1=16, dim0_stride=65536}.
__device__ __forceinline__ void tdm_load_tile(unsigned lds_off,
                                              unsigned long long gaddr)
{
    v4u g0;
    g0[0] = 1u;                                   // count=1, user descriptor
    g0[1] = lds_off;                              // LDS byte address
    g0[2] = (unsigned)gaddr;                      // global addr [31:0]
    g0[3] = (unsigned)((gaddr >> 32) & 0x1FFFFFFull) | (2u << 30); // [56:32]|type=2

    v8i_ g1;
    g1[0] = 0x20000;                              // data_size = 2 (4 bytes)
    g1[1] = (S_DIM << 16);                        // tensor_dim0 = 64
    g1[2] = (TILE_B << 16);                       // tensor_dim1 = 16
    g1[3] = (S_DIM << 16);                        // tile_dim0 = 64
    g1[4] = TILE_B;                               // tile_dim1 = 16, tile_dim2 = 0
    g1[5] = T_DIM * S_DIM;                        // tensor_dim0_stride = 65536
    g1[6] = 0;
    g1[7] = 0;

    v4i z4 = {0, 0, 0, 0};
#if __clang_major__ >= 23
    v8i_ z8 = {0, 0, 0, 0, 0, 0, 0, 0};
    __builtin_amdgcn_tensor_load_to_lds(g0, g1, z4, z4, z8, 0);
#else
    __builtin_amdgcn_tensor_load_to_lds(g0, g1, z4, z4, 0);
#endif
}

__device__ __forceinline__ unsigned lds_offset_of(const void* p)
{
    // generic pointer to __shared__: low 32 bits are the wave-relative LDS
    // byte offset (ISA 10.2 aperture mapping)
    return (unsigned)(unsigned long long)(uintptr_t)p;
}

__global__ __launch_bounds__(WAVES * 32, 1)
void snn_lif_wmma_kernel(const float* __restrict__ sensor,
                         const float* __restrict__ noise,
                         const float* __restrict__ Wh,
                         const float* __restrict__ bh,
                         const float* __restrict__ Wo,
                         const float* __restrict__ bo,
                         float* __restrict__ out)
{
    // double-buffered TDM staging (16x64 f32 each), packed A-fragments, partials
    __shared__ __align__(16) float stage_s[2][TILE_B * S_DIM];
    __shared__ __align__(16) float stage_n[2][TILE_B * S_DIM];
    __shared__ v8h   Apack[2][128];               // [buf][(kk*32+lane)*2 + half]
    __shared__ float red[2][WAVES][TILE_B];

    const int tid   = threadIdx.x;
    const int lane  = tid & 31;
    const int wave  = tid >> 5;
    const int n16   = lane & 15;
    const int hi    = lane >> 4;
    const int b0    = blockIdx.x * TILE_B;
    const int hbase = wave * (NT * 16);

    // ---- preload W_h^T slice as f16 WMMA B-tiles, resident in VGPRs ----
    v16h bmat[NT][2];
#pragma unroll
    for (int nt = 0; nt < NT; ++nt) {
        const float* wr = Wh + (hbase + nt * 16 + n16) * S_DIM;
#pragma unroll
        for (int kk = 0; kk < 2; ++kk) {
            const int kb = kk * 32 + hi * 16;
            const fvec4 w0 = *(const fvec4*)(wr + kb + 0);
            const fvec4 w1 = *(const fvec4*)(wr + kb + 4);
            const fvec4 w2 = *(const fvec4*)(wr + kb + 8);
            const fvec4 w3 = *(const fvec4*)(wr + kb + 12);
            v16h b;
#pragma unroll
            for (int e = 0; e < 4; ++e) b[e]      = (_Float16)w0[e];
#pragma unroll
            for (int e = 0; e < 4; ++e) b[4 + e]  = (_Float16)w1[e];
#pragma unroll
            for (int e = 0; e < 4; ++e) b[8 + e]  = (_Float16)w2[e];
#pragma unroll
            for (int e = 0; e < 4; ++e) b[12 + e] = (_Float16)w3[e];
            bmat[nt][kk] = b;
        }
    }

    float bh_v[NT], wo_v[NT];
#pragma unroll
    for (int nt = 0; nt < NT; ++nt) {
        bh_v[nt] = bh[hbase + nt * 16 + n16];
        wo_v[nt] = Wo[hbase + nt * 16 + n16];
    }
    const float b_o = bo[0];

    v8f vh[NT];
#pragma unroll
    for (int nt = 0; nt < NT; ++nt)
#pragma unroll
        for (int e = 0; e < 8; ++e) vh[nt][e] = 0.0f;

    float vo = 0.0f, acc = 0.0f;

    // packer role (first 128 threads): half A-fragment = 8 contiguous K of a row
    const bool packer = (tid < 128);
    const int pkk  = tid >> 6;          // tile (K block of 32)
    const int pln  = (tid >> 1) & 31;   // destination fragment lane
    const int phf  = tid & 1;           // which half of the v16h
    const int prow = pln & 15;
    const int pK   = pkk * 32 + (pln >> 4) * 16 + phf * 8;

    const unsigned long long sens_g =
        (unsigned long long)(uintptr_t)sensor + (unsigned long long)b0 * T_DIM * S_DIM * 4ull;
    const unsigned long long nois_g =
        (unsigned long long)(uintptr_t)noise  + (unsigned long long)b0 * T_DIM * S_DIM * 4ull;

    // prologue: stage t = 0 (TDM issue split across two waves)
    if (wave == 0) tdm_load_tile(lds_offset_of(&stage_s[0][0]), sens_g);
    if (wave == 1) tdm_load_tile(lds_offset_of(&stage_n[0][0]), nois_g);

    for (int t = 0; t < T_DIM; ++t) {
        const int buf = t & 1;

        if (wave < 2) __builtin_amdgcn_s_wait_tensorcnt(0);
        __syncthreads();                            // staged f32 tiles visible

        if ((t + 1) < T_DIM) {                      // overlap next TDM with compute
            const unsigned long long off = (unsigned long long)(t + 1) * S_DIM * 4ull;
            if (wave == 0) tdm_load_tile(lds_offset_of(&stage_s[buf ^ 1][0]), sens_g + off);
            if (wave == 1) tdm_load_tile(lds_offset_of(&stage_n[buf ^ 1][0]), nois_g + off);
        }

        // ---- cooperative spike pack: done ONCE per block, WMMA A layout ----
        if (packer) {
            const fvec4 sA = *(const fvec4*)&stage_s[buf][prow * S_DIM + pK];
            const fvec4 sB = *(const fvec4*)&stage_s[buf][prow * S_DIM + pK + 4];
            const fvec4 nA = *(const fvec4*)&stage_n[buf][prow * S_DIM + pK];
            const fvec4 nB = *(const fvec4*)&stage_n[buf][prow * S_DIM + pK + 4];
            v8h u;
#pragma unroll
            for (int e = 0; e < 4; ++e) u[e]     = (_Float16)((nA[e] < sA[e]) ? 1.0f : 0.0f);
#pragma unroll
            for (int e = 0; e < 4; ++e) u[4 + e] = (_Float16)((nB[e] < sB[e]) ? 1.0f : 0.0f);
            Apack[buf][(pkk * 32 + pln) * 2 + phf] = u;
        }
        __syncthreads();                            // A-fragments ready

        // ---- fetch shared A-fragments (identical for all 8 waves) ----
        v16h a[2];
#pragma unroll
        for (int kk = 0; kk < 2; ++kk) {
            union { v8h h[2]; v16h v; } ua;
            ua.h[0] = Apack[buf][(kk * 32 + lane) * 2 + 0];
            ua.h[1] = Apack[buf][(kk * 32 + lane) * 2 + 1];
            a[kk] = ua.v;
        }

        // ---- hidden LIF: v_h = beta*v_h + spikes @ Wh^T + b_h, thresh, reset ----
        float p[8];
#pragma unroll
        for (int e = 0; e < 8; ++e) p[e] = 0.0f;

#pragma unroll
        for (int nt = 0; nt < NT; ++nt) {
            v8f c;
#pragma unroll
            for (int e = 0; e < 8; ++e) c[e] = BETA_H * vh[nt][e] + bh_v[nt];
            v8f d = __builtin_amdgcn_wmma_f32_16x16x32_f16(
                        false, a[0], false, bmat[nt][0], (short)0, c, false, false);
            d = __builtin_amdgcn_wmma_f32_16x16x32_f16(
                        false, a[1], false, bmat[nt][1], (short)0, d, false, false);
#pragma unroll
            for (int e = 0; e < 8; ++e) {
                const bool s = (d[e] >= 1.0f);
                p[e] += s ? wo_v[nt] : 0.0f;        // s_h * W_o contribution
                vh[nt][e] = s ? 0.0f : d[e];        // hard reset
            }
        }

        // reduce readout partials across the 16 columns in this lane group
#pragma unroll
        for (int off = 1; off < 16; off <<= 1)
#pragma unroll
            for (int e = 0; e < 8; ++e) p[e] += __shfl_xor(p[e], off, 16);

        if (lane == 0) {
#pragma unroll
            for (int e = 0; e < 8; ++e) red[buf][wave][e] = p[e];      // rows 0..7
        }
        if (lane == 16) {
#pragma unroll
            for (int e = 0; e < 8; ++e) red[buf][wave][8 + e] = p[e];  // rows 8..15
        }
        __syncthreads();

        // ---- output LIF (wave 0, one batch row per lane slot) ----
        if (wave == 0) {
            float io = red[buf][0][n16];
#pragma unroll
            for (int w = 1; w < WAVES; ++w) io += red[buf][w][n16];
            vo = BETA_O * vo + io + b_o;
            const float so = (vo >= 1.0f) ? 1.0f : 0.0f;
            acc += so;
            vo = (so != 0.0f) ? 0.0f : vo;
        }
    }

    if (wave == 0 && lane < 16) {
        out[b0 + n16] = acc * (1.0f / (float)T_DIM);
    }
}

extern "C" void kernel_launch(void* const* d_in, const int* in_sizes, int n_in,
                              void* d_out, int out_size, void* d_ws, size_t ws_size,
                              hipStream_t stream) {
    (void)in_sizes; (void)n_in; (void)d_ws; (void)ws_size; (void)out_size;
    const float* sensor = (const float*)d_in[0];
    const float* noise  = (const float*)d_in[1];
    const float* Wh     = (const float*)d_in[2];
    const float* bh     = (const float*)d_in[3];
    const float* Wo     = (const float*)d_in[4];
    const float* bo     = (const float*)d_in[5];
    float* out = (float*)d_out;

    dim3 grid(B_DIM / TILE_B);   // 128 blocks
    dim3 block(WAVES * 32);      // 8 waves -> 2 waves/SIMD on a 128-WGP part
    hipLaunchKernelGGL(snn_lif_wmma_kernel, grid, block, 0, stream,
                       sensor, noise, Wh, bh, Wo, bo, out);
}
